// PELICANClassifier_19061064859902
// MI455X (gfx1250) — compile-verified, matrix operand order checked by reading, not run on hardware
//
#include <hip/hip_runtime.h>
#include <hip/hip_fp16.h>

// ---------------- problem constants ----------------
#define BB 32
#define NN 64
#define SS 9
#define CC 128
#define CS 32
#define C2 96
#define NL 4
#define AVGK 49.0f
#define NPB 4096            // N*N rows per batch
#define NROW (BB * NPB)     // total rows = 131072
#define WMAT (CC * CC)      // 16384 elements per 128x128 matrix

typedef _Float16 h8  __attribute__((ext_vector_type(8)));
typedef _Float16 h16 __attribute__((ext_vector_type(16)));
typedef float    f8  __attribute__((ext_vector_type(8)));

__device__ __forceinline__ float lrelu(float x) { return x > 0.0f ? x : 0.01f * x; }

// ---- gfx1250 async global->LDS copy (16B per lane), ASYNCcnt tracked ----
__device__ __forceinline__ void async_copy16(const _Float16* g, _Float16* lds_generic) {
  unsigned ldsAddr = (unsigned)(size_t)lds_generic;             // addr[31:0] = LDS offset
  unsigned long long ga = (unsigned long long)(size_t)g;
  asm volatile("global_load_async_to_lds_b128 %0, %1, off"
               :: "v"(ldsAddr), "v"(ga) : "memory");
}
__device__ __forceinline__ void wait_async0() {
  asm volatile("s_wait_asynccnt 0" ::: "memory");
}

// ---- WMMA fragment loads (CDNA5 16x16x32 f16 layouts, ISA 7.12.2) ----
// A (16x32, MxK): lane l holds row (l&15); halves 0..7 -> K = kb+(l>>4)*8 .. +7,
// halves 8..15 -> +16.  Two contiguous 16B loads per lane.
__device__ __forceinline__ h16 load_fragA(const _Float16* __restrict__ mat,
                                          long row, int lane, int kb) {
  int ko = (lane >> 4) * 8;
  const _Float16* p = mat + row * (long)CC + kb + ko;
  h8 lo = *(const h8*)p;
  h8 hi = *(const h8*)(p + 16);
  h16 o;
#pragma unroll
  for (int t = 0; t < 8; ++t) { o[t] = lo[t]; o[8 + t] = hi[t]; }
  return o;
}
// B (32x16, KxN) from pre-transposed weights Wt[n][k] staged in LDS: lane l holds
// col (l&15), same K striping as A -> contiguous 16B ds loads.
__device__ __forceinline__ h16 load_fragB_lds(const _Float16* Wl, int n0, int lane, int kb) {
  int n  = n0 + (lane & 15);
  int ko = (lane >> 4) * 8;
  const _Float16* p = Wl + n * CC + kb + ko;
  h8 lo = *(const h8*)p;
  h8 hi = *(const h8*)(p + 16);
  h16 o;
#pragma unroll
  for (int t = 0; t < 8; ++t) { o[t] = lo[t]; o[8 + t] = hi[t]; }
  return o;
}

// ---------------- weight pack: f32 [k][n] -> f16 transposed [n][k] ----------------
__global__ __launch_bounds__(128) void k_packW(const float* __restrict__ src,
                                               _Float16* __restrict__ dst) {
  int n = blockIdx.x, k = threadIdx.x;
  dst[n * CC + k] = (_Float16)src[k * CC + n];
}

// ---------------- input prep: mask, eq1to2 precursors ----------------
__global__ __launch_bounds__(256) void k_pre0(const float* __restrict__ scalars,
                                              const int* __restrict__ nobj,
                                              const float* __restrict__ w_in,
                                              float* __restrict__ mask1,
                                              float* __restrict__ A0, float* __restrict__ A1,
                                              float* __restrict__ A2,
                                              float* __restrict__ S3, float* __restrict__ S4) {
  int b = blockIdx.x, t = threadIdx.x;
  __shared__ float sm[NN];
  __shared__ float xs[SS];
  if (t < NN) { float m = (t < nobj[b]) ? 1.0f : 0.0f; sm[t] = m; mask1[b * NN + t] = m; }
  __syncthreads();
  if (t < SS) {
    float s = 0.0f;
    for (int i = 0; i < NN; ++i) s += scalars[(b * NN + i) * SS + t] * sm[i];
    xs[t] = s / AVGK;
  }
  __syncthreads();
  for (int idx = t; idx < NN * CS; idx += 256) {
    int i = idx >> 5, c = idx & 31;
    float mi = sm[i], a0 = 0.0f, a1 = 0.0f, a2 = 0.0f;
#pragma unroll
    for (int s = 0; s < SS; ++s) {
      float x = scalars[(b * NN + i) * SS + s] * mi;
      a0 += x * w_in[0 * SS * CS + s * CS + c];
      a1 += x * w_in[1 * SS * CS + s * CS + c];
      a2 += x * w_in[2 * SS * CS + s * CS + c];
    }
    A0[(b * NN + i) * CS + c] = a0;
    A1[(b * NN + i) * CS + c] = a1;
    A2[(b * NN + i) * CS + c] = a2;
  }
  if (t < CS) {
    float s3 = 0.0f, s4 = 0.0f;
#pragma unroll
    for (int s = 0; s < SS; ++s) {
      s3 += xs[s] * w_in[3 * SS * CS + s * CS + t];
      s4 += xs[s] * w_in[4 * SS * CS + s * CS + t];
    }
    S3[b * CS + t] = s3; S4[b * CS + t] = s4;
  }
}

// ---------------- build T0 = concat(eq1to2, slog(Minkowski)) in f16 ----------------
__global__ __launch_bounds__(128) void k_build(const float* __restrict__ momenta,
                                               const float* __restrict__ w_lin,
                                               const float* __restrict__ alpha,
                                               const float* __restrict__ b_in,
                                               const float* __restrict__ mask1,
                                               const float* __restrict__ A0,
                                               const float* __restrict__ A1,
                                               const float* __restrict__ A2,
                                               const float* __restrict__ S3,
                                               const float* __restrict__ S4,
                                               _Float16* __restrict__ T0) {
  int blk = blockIdx.x;
  int b = blk >> 12, p = blk & (NPB - 1);
  int i = p >> 6, j = p & 63;
  int c = threadIdx.x;
  __shared__ float dot;
  if (c == 0) {
    const float* pi = momenta + (b * NN + i) * 4;
    const float* pj = momenta + (b * NN + j) * 4;
    dot = pi[0] * pj[0] - pi[1] * pj[1] - pi[2] * pj[2] - pi[3] * pj[3];
  }
  float mk = mask1[b * NN + i] * mask1[b * NN + j];
  __syncthreads();
  float v;
  if (c < CS) {
    v = A0[(b * NN + i) * CS + c] + A1[(b * NN + j) * CS + c] + S3[b * CS + c] + b_in[c];
    if (i == j) v += A2[(b * NN + i) * CS + c] + S4[b * CS + c];
    v = lrelu(v) * mk;
  } else {
    int c2 = c - CS;
    float r2 = dot * w_lin[c2];
    v = alpha[c2] * copysignf(log1pf(fabsf(r2)), r2) * mk;
  }
  T0[(long)blk * CC + c] = (_Float16)v;
}

// ---------------- MessageNet: U = lrelu(T @ W + b) * mask  (WMMA) ----------------
__global__ __launch_bounds__(256) void k_gemm_msg(const _Float16* __restrict__ Tin,
                                                  const _Float16* __restrict__ Wt,
                                                  const float* __restrict__ bias,
                                                  const float* __restrict__ mask1,
                                                  _Float16* __restrict__ Uout) {
  __shared__ _Float16 sW[WMAT];                 // 32KB: weights staged via async DMA
  for (int idx = threadIdx.x; idx < WMAT / 8; idx += 256)
    async_copy16(Wt + idx * 8, &sW[idx * 8]);
  wait_async0();
  __syncthreads();

  int wave = threadIdx.x >> 5, lane = threadIdx.x & 31;
  long rowBase = (long)blockIdx.x * 128 + wave * 16;
  f8 acc[8] = {};
#pragma unroll
  for (int kb = 0; kb < CC; kb += 32) {
    h16 a = load_fragA(Tin, rowBase + (lane & 15), lane, kb);
#pragma unroll
    for (int nt = 0; nt < 8; ++nt) {
      h16 bf = load_fragB_lds(sW, nt * 16, lane, kb);
      acc[nt] = __builtin_amdgcn_wmma_f32_16x16x32_f16(false, a, false, bf,
                                                       (short)0, acc[nt], false, false);
    }
  }
  int nlo = lane & 15, mhalf = (lane >> 4) * 8;
  long bIdx = rowBase >> 12;
  const float* mrow = mask1 + bIdx * NN;
#pragma unroll
  for (int nt = 0; nt < 8; ++nt) {
    int n = nt * 16 + nlo;
    float bn = bias[n];
#pragma unroll
    for (int r = 0; r < 8; ++r) {
      long row = rowBase + mhalf + r;
      int p = (int)(row & (NPB - 1));
      int i = p >> 6, j = p & 63;
      float v = lrelu(acc[nt][r] + bn) * mrow[i] * mrow[j];
      Uout[row * CC + n] = (_Float16)v;
    }
  }
}

// ---------------- reductions over U: diag, row sums, col sums ----------------
__global__ __launch_bounds__(128) void k_reduce(const _Float16* __restrict__ U,
                                                float* __restrict__ red_r,
                                                float* __restrict__ red_c,
                                                float* __restrict__ red_d) {
  int c = threadIdx.x, bn = blockIdx.x;
  int b = bn >> 6, n = bn & 63;
  const _Float16* br = U + ((long)b * NPB + n * 64) * CC + c;
  float s = 0.0f;
  for (int j = 0; j < NN; ++j) s += (float)br[j * CC];
  red_r[bn * CC + c] = s * (1.0f / AVGK);
  const _Float16* bc = U + ((long)b * NPB + n) * CC + c;
  float sc = 0.0f;
  for (int i = 0; i < NN; ++i) sc += (float)bc[(long)i * 64 * CC];
  red_c[bn * CC + c] = sc * (1.0f / AVGK);
  red_d[bn * CC + c] = (float)U[((long)b * NPB + n * 65) * CC + c];
}

// ---------------- aux: Arow/Acol/Adiag = [d r cl] against eq_w slices ----------------
__global__ __launch_bounds__(128) void k_aux(const float* __restrict__ red_r,
                                             const float* __restrict__ red_c,
                                             const float* __restrict__ red_d,
                                             const float* __restrict__ W,
                                             float* __restrict__ Arow,
                                             float* __restrict__ Acol,
                                             float* __restrict__ Adiag) {
  int c = threadIdx.x, bn = blockIdx.x;
  __shared__ float sd[CC], sr[CC], scl[CC];
  sd[c] = red_d[bn * CC + c]; sr[c] = red_r[bn * CC + c]; scl[c] = red_c[bn * CC + c];
  __syncthreads();
  float ar = 0.0f, ac = 0.0f, ad = 0.0f;
  for (int k = 0; k < CC; ++k) {
    float dk = sd[k], rk = sr[k], ck = scl[k];
    ad += dk * W[2 * WMAT + k * CC + c] + rk * W[9 * WMAT + k * CC + c] + ck * W[10 * WMAT + k * CC + c];
    ar += dk * W[3 * WMAT + k * CC + c] + rk * W[5 * WMAT + k * CC + c] + ck * W[7 * WMAT + k * CC + c];
    ac += dk * W[4 * WMAT + k * CC + c] + rk * W[6 * WMAT + k * CC + c] + ck * W[8 * WMAT + k * CC + c];
  }
  Arow[bn * CC + c] = ar; Acol[bn * CC + c] = ac; Adiag[bn * CC + c] = ad;
}

// ---------------- aux-b: Ab/Abd from totals/trace + biases ----------------
__global__ __launch_bounds__(128) void k_auxb(const float* __restrict__ red_r,
                                              const float* __restrict__ red_d,
                                              const float* __restrict__ W,
                                              const float* __restrict__ eqb,
                                              const float* __restrict__ eqbd,
                                              float* __restrict__ Ab, float* __restrict__ Abd) {
  int c = threadIdx.x, b = blockIdx.x;
  __shared__ float st[CC], str_[CC];
  float tt = 0.0f, trr = 0.0f;
  for (int i = 0; i < NN; ++i) {
    tt  += red_r[(b * NN + i) * CC + c];
    trr += red_d[(b * NN + i) * CC + c];
  }
  st[c] = tt / AVGK; str_[c] = trr / AVGK;
  __syncthreads();
  float ab = eqb[c], abd = eqbd[c];
  for (int k = 0; k < CC; ++k) {
    ab  += st[k] * W[11 * WMAT + k * CC + c] + str_[k] * W[13 * WMAT + k * CC + c];
    abd += st[k] * W[12 * WMAT + k * CC + c] + str_[k] * W[14 * WMAT + k * CC + c];
  }
  Ab[b * CC + c] = ab; Abd[b * CC + c] = abd;
}

// ---------------- Eq2to2 main: T' = lrelu(U@W0 + Ut@W1 + broadcasts) * mask ----------------
__global__ __launch_bounds__(256) void k_gemm_eq2to2(const _Float16* __restrict__ U,
                                                     const _Float16* __restrict__ Wt0,
                                                     const _Float16* __restrict__ Wt1,
                                                     const float* __restrict__ Arow,
                                                     const float* __restrict__ Acol,
                                                     const float* __restrict__ Adiag,
                                                     const float* __restrict__ Ab,
                                                     const float* __restrict__ Abd,
                                                     const float* __restrict__ mask1,
                                                     _Float16* __restrict__ Tout) {
  __shared__ _Float16 sW0[WMAT];                // 64KB total: both weight mats via async DMA
  __shared__ _Float16 sW1[WMAT];
  for (int idx = threadIdx.x; idx < WMAT / 8; idx += 256) {
    async_copy16(Wt0 + idx * 8, &sW0[idx * 8]);
    async_copy16(Wt1 + idx * 8, &sW1[idx * 8]);
  }
  wait_async0();
  __syncthreads();

  int wave = threadIdx.x >> 5, lane = threadIdx.x & 31;
  long rowBase = (long)blockIdx.x * 128 + wave * 16;
  long bIdx = rowBase >> 12;
  int p0 = (int)(rowBase & (NPB - 1));
  int i0 = p0 >> 6, j0 = p0 & 63;          // strip: i fixed, j = j0..j0+15
  f8 acc[8] = {};
  // direct part: U @ W0
#pragma unroll
  for (int kb = 0; kb < CC; kb += 32) {
    h16 a = load_fragA(U, rowBase + (lane & 15), lane, kb);
#pragma unroll
    for (int nt = 0; nt < 8; ++nt) {
      h16 bf = load_fragB_lds(sW0, nt * 16, lane, kb);
      acc[nt] = __builtin_amdgcn_wmma_f32_16x16x32_f16(false, a, false, bf,
                                                       (short)0, acc[nt], false, false);
    }
  }
  // transposed part: row m needs U[b, j0+m, i0, :]
#pragma unroll
  for (int kb = 0; kb < CC; kb += 32) {
    long rT = (bIdx << 12) + (long)(j0 + (lane & 15)) * 64 + i0;
    h16 a = load_fragA(U, rT, lane, kb);
#pragma unroll
    for (int nt = 0; nt < 8; ++nt) {
      h16 bf = load_fragB_lds(sW1, nt * 16, lane, kb);
      acc[nt] = __builtin_amdgcn_wmma_f32_16x16x32_f16(false, a, false, bf,
                                                       (short)0, acc[nt], false, false);
    }
  }
  int nlo = lane & 15, mhalf = (lane >> 4) * 8;
  const float* mrow = mask1 + bIdx * NN;
  float mi = mrow[i0];
#pragma unroll
  for (int nt = 0; nt < 8; ++nt) {
    int n = nt * 16 + nlo;
    float ar  = Arow[(bIdx * NN + i0) * CC + n];
    float ad  = Adiag[(bIdx * NN + i0) * CC + n];
    float abv = Ab[bIdx * CC + n];
    float abd = Abd[bIdx * CC + n];
#pragma unroll
    for (int r = 0; r < 8; ++r) {
      int j = j0 + mhalf + r;
      float v = acc[nt][r] + ar + Acol[(bIdx * NN + j) * CC + n] + abv;
      if (j == i0) v += ad + abd;
      v = lrelu(v) * mi * mrow[j];
      long row = rowBase + mhalf + r;
      Tout[row * CC + n] = (_Float16)v;
    }
  }
}

// ---------------- Eq2to0 + output MLP ----------------
__global__ __launch_bounds__(128) void k_final(const float* __restrict__ red_r,
                                               const float* __restrict__ red_d,
                                               const float* __restrict__ w20,
                                               const float* __restrict__ b20,
                                               const float* __restrict__ wmlp,
                                               const float* __restrict__ bmlp,
                                               float* __restrict__ out) {
  int b = blockIdx.x, d = threadIdx.x;
  __shared__ float ag0[CC], ag1[CC], sh[CC];
  float tt = 0.0f, trr = 0.0f;
  for (int i = 0; i < NN; ++i) {
    tt  += red_r[(b * NN + i) * CC + d];
    trr += red_d[(b * NN + i) * CC + d];
  }
  ag0[d] = lrelu(trr / AVGK);   // dg (trace)
  ag1[d] = lrelu(tt / AVGK);    // tot
  __syncthreads();
  float act = b20[d];
  for (int c = 0; c < CC; ++c)
    act += ag0[c] * w20[c * CC + d] + ag1[c] * w20[WMAT + c * CC + d];
  for (int kk = 0; kk < 2; ++kk) {
    sh[d] = act * wmlp[d * 2 + kk];
    __syncthreads();
    for (int s = 64; s > 0; s >>= 1) { if (d < s) sh[d] += sh[d + s]; __syncthreads(); }
    if (d == 0) out[b * 2 + kk] = sh[0] + bmlp[kk];
    __syncthreads();
  }
}

// ---------------- host launcher ----------------
extern "C" void kernel_launch(void* const* d_in, const int* in_sizes, int n_in,
                              void* d_out, int out_size, void* d_ws, size_t ws_size,
                              hipStream_t stream) {
  (void)in_sizes; (void)out_size; (void)ws_size;
  const float* momenta = (const float*)d_in[0];
  const float* scalars = (const float*)d_in[1];
  const int*   nobj    = (const int*)d_in[2];
  const float* w_lin   = (const float*)d_in[3];
  const float* alpha   = (const float*)d_in[4];
  const float* w_in    = (const float*)d_in[5];
  const float* b_in    = (const float*)d_in[6];
  const float *msg_w[NL], *msg_b[NL], *eq_w[NL], *eq_b[NL], *eq_bd[NL];
  const float *wm0, *bm0, *w20, *b20, *wmlp, *bmlp;
  if (n_in >= 30) {  // tuple entries passed as separate inputs
    for (int l = 0; l < NL; ++l) {
      msg_w[l] = (const float*)d_in[7 + l];
      msg_b[l] = (const float*)d_in[11 + l];
      eq_w[l]  = (const float*)d_in[15 + l];
      eq_b[l]  = (const float*)d_in[19 + l];
      eq_bd[l] = (const float*)d_in[23 + l];
    }
    wm0 = (const float*)d_in[27]; bm0 = (const float*)d_in[28];
    w20 = (const float*)d_in[29]; b20 = (const float*)d_in[30];
    wmlp = (const float*)d_in[31]; bmlp = (const float*)d_in[32];
  } else {           // tuples flattened into single arrays
    for (int l = 0; l < NL; ++l) {
      msg_w[l] = (const float*)d_in[7] + (size_t)l * WMAT;
      msg_b[l] = (const float*)d_in[8] + (size_t)l * CC;
      eq_w[l]  = (const float*)d_in[9] + (size_t)l * 15 * WMAT;
      eq_b[l]  = (const float*)d_in[10] + (size_t)l * CC;
      eq_bd[l] = (const float*)d_in[11] + (size_t)l * CC;
    }
    wm0 = (const float*)d_in[12]; bm0 = (const float*)d_in[13];
    w20 = (const float*)d_in[14]; b20 = (const float*)d_in[15];
    wmlp = (const float*)d_in[16]; bmlp = (const float*)d_in[17];
  }

  // workspace carve-up (256B aligned)
  unsigned char* w = (unsigned char*)d_ws;
  size_t off = 0;
  auto take = [&](size_t nbytes) -> void* {
    void* p = w + off;
    off = (off + nbytes + 255) & ~(size_t)255;
    return p;
  };
  float* mask1 = (float*)take(BB * NN * 4);
  float* A0 = (float*)take(BB * NN * CS * 4);
  float* A1 = (float*)take(BB * NN * CS * 4);
  float* A2 = (float*)take(BB * NN * CS * 4);
  float* S3 = (float*)take(BB * CS * 4);
  float* S4 = (float*)take(BB * CS * 4);
  _Float16* Wt = (_Float16*)take((size_t)13 * WMAT * 2);
  float* red_r = (float*)take((size_t)BB * NN * CC * 4);
  float* red_c = (float*)take((size_t)BB * NN * CC * 4);
  float* red_d = (float*)take((size_t)BB * NN * CC * 4);
  float* Arow  = (float*)take((size_t)BB * NN * CC * 4);
  float* Acol  = (float*)take((size_t)BB * NN * CC * 4);
  float* Adiag = (float*)take((size_t)BB * NN * CC * 4);
  float* Ab  = (float*)take(BB * CC * 4);
  float* Abd = (float*)take(BB * CC * 4);
  _Float16* T = (_Float16*)take((size_t)NROW * CC * 2);
  _Float16* U = (_Float16*)take((size_t)NROW * CC * 2);

  _Float16* Wt_msg0 = Wt;                 // slots 0..3
  _Float16* Wt_eq0  = Wt + 4 * WMAT;      // slots 4..7
  _Float16* Wt_eq1  = Wt + 8 * WMAT;      // slots 8..11
  _Float16* Wt_m0   = Wt + 12 * WMAT;     // slot 12

  // pack weights (f32 [k][n] -> f16 [n][k])
  for (int l = 0; l < NL; ++l) {
    k_packW<<<CC, CC, 0, stream>>>(msg_w[l], Wt_msg0 + l * WMAT);
    k_packW<<<CC, CC, 0, stream>>>(eq_w[l] + 0 * WMAT, Wt_eq0 + l * WMAT);
    k_packW<<<CC, CC, 0, stream>>>(eq_w[l] + 1 * WMAT, Wt_eq1 + l * WMAT);
  }
  k_packW<<<CC, CC, 0, stream>>>(wm0, Wt_m0);

  // input stage
  k_pre0<<<BB, 256, 0, stream>>>(scalars, nobj, w_in, mask1, A0, A1, A2, S3, S4);
  k_build<<<BB * NPB, CC, 0, stream>>>(momenta, w_lin, alpha, b_in, mask1,
                                       A0, A1, A2, S3, S4, T);

  // Net2to2 layers
  for (int l = 0; l < NL; ++l) {
    k_gemm_msg<<<NROW / 128, 256, 0, stream>>>(T, Wt_msg0 + l * WMAT, msg_b[l], mask1, U);
    k_reduce<<<BB * NN, CC, 0, stream>>>(U, red_r, red_c, red_d);
    k_aux<<<BB * NN, CC, 0, stream>>>(red_r, red_c, red_d, eq_w[l], Arow, Acol, Adiag);
    k_auxb<<<BB, CC, 0, stream>>>(red_r, red_d, eq_w[l], eq_b[l], eq_bd[l], Ab, Abd);
    k_gemm_eq2to2<<<NROW / 128, 256, 0, stream>>>(U, Wt_eq0 + l * WMAT, Wt_eq1 + l * WMAT,
                                                  Arow, Acol, Adiag, Ab, Abd, mask1, T);
  }

  // msg_2to0 + Eq2to0 + MLP head
  k_gemm_msg<<<NROW / 128, 256, 0, stream>>>(T, Wt_m0, bm0, mask1, U);
  k_reduce<<<BB * NN, CC, 0, stream>>>(U, red_r, red_c, red_d);
  k_final<<<BB, CC, 0, stream>>>(red_r, red_d, w20, b20, wmlp, bmlp, (float*)d_out);
}
